// MambaModel_56642028700213
// MI455X (gfx1250) — matrix operand order repaired and checked by reading, not compiled, not run on hardware
//
#include <hip/hip_runtime.h>
#include <hip/hip_bf16.h>

#define BATCH   4
#define SEQLEN  2048
#define DM      256
#define DI      512
#define DSTATE  16
#define DTRANK  16
#define BS      (BATCH * SEQLEN)   // 8192 token rows
#define LAYERS  4
#define NXP     64                 // x-proj N padded 48 -> 64 (stride of dbl)

typedef __attribute__((ext_vector_type(16))) _Float16 v16h;
typedef __attribute__((ext_vector_type(8)))  _Float16 v8h;
typedef __attribute__((ext_vector_type(8)))  float    v8f;

// ---------------------------------------------------------------------------
// WMMA fragment helpers (CDNA5 16x16x32 f16, wave32)
// A: row-major M x K.  ISA layout: lanes 0-15 row m=lane, VGPR0-3 = K 0..7,
// VGPR4-7 = K 16..23; lanes 16-31 same rows, K 8..15 / 24..31.
// ---------------------------------------------------------------------------
__device__ __forceinline__ v16h frag_load_A(const _Float16* __restrict__ A,
                                            int lda, int m0, int k0, int lane) {
    int m  = m0 + (lane & 15);
    int kb = k0 + ((lane >> 4) << 3);          // +0 or +8
    const _Float16* p = A + (size_t)m * lda + kb;
    v8h lo = *(const v8h*)p;                   // K = kb .. kb+7
    v8h hi = *(const v8h*)(p + 16);            // K = kb+16 .. kb+23
    v16h r;
#pragma unroll
    for (int i = 0; i < 8; ++i) { r[i] = lo[i]; r[i + 8] = hi[i]; }
    return r;
}

// B operand supplied transposed: Bw is row-major (N x K) = B^T, so a fragment
// column n is a contiguous K-run.  Layout (matches A): lanes 0-15 hold col
// n=lane, K 0..7 / 16..23; lanes 16-31 same cols, K 8..15 / 24..31.
__device__ __forceinline__ v16h frag_load_B(const _Float16* __restrict__ Bw,
                                            int ldb, int n0, int k0, int lane) {
    int n  = n0 + (lane & 15);
    int kb = k0 + ((lane >> 4) << 3);          // +0 or +8
    const _Float16* p = Bw + (size_t)n * ldb + kb;
    v8h lo = *(const v8h*)p;
    v8h hi = *(const v8h*)(p + 16);
    v16h r;
#pragma unroll
    for (int i = 0; i < 8; ++i) { r[i] = lo[i]; r[i + 8] = hi[i]; }
    return r;
}

// C/D 16x16 f32: VGPR e, lanes 0-15 -> M=e, lanes 16-31 -> M=e+8; N = lane&15.
__device__ __forceinline__ void frag_store_D(float* __restrict__ C, int ldc,
                                             int m0, int n0, int lane, v8f acc) {
    int n  = n0 + (lane & 15);
    int mb = m0 + ((lane >> 4) << 3);
#pragma unroll
    for (int e = 0; e < 8; ++e)
        C[(size_t)(mb + e) * ldc + n] = acc[e];
}

// ---------------------------------------------------------------------------
// GEMM: C(MxN,f32) = A(MxK,f16 row-major) * Bw(NxK,f16 row-major)^T
// One wave computes a 32x64 macro-tile (2 M-tiles x 4 N-tiles, 8 wmma/K-step).
// Software-pipelined: A fragments prefetched one K-step ahead, B fragments one
// N-tile ahead, so wmma bursts overlap outstanding global_load_b128s.
// Requirements: M % 32 == 0, N % 64 == 0, K % 32 == 0.
// grid = (N / (64 * wavesPerBlock), M / 32), block = 32 * wavesPerBlock.
// Branch-free: EXEC stays all-1 through every wmma.
// ---------------------------------------------------------------------------
__global__ void __launch_bounds__(128)
wmma_gemm_nt(const _Float16* __restrict__ A,
             const _Float16* __restrict__ Bw,
             float* __restrict__ C, int M, int N, int K) {
    int lane = threadIdx.x & 31;
    int wave = threadIdx.x >> 5;
    int m0 = blockIdx.y << 5;                                  // 32 rows / block
    int n0 = (blockIdx.x * (blockDim.x >> 5) + wave) << 6;     // 64 cols / wave

    v8f acc[2][4];
#pragma unroll
    for (int i = 0; i < 2; ++i)
#pragma unroll
        for (int j = 0; j < 4; ++j) acc[i][j] = v8f{};

    // Prologue: fragments for k0 = 0, first B tile
    v16h a0 = frag_load_A(A, K, m0,      0, lane);
    v16h a1 = frag_load_A(A, K, m0 + 16, 0, lane);
    v16h b  = frag_load_B(Bw, K, n0, 0, lane);

    for (int k0 = 0; k0 < K; k0 += 32) {
        int k1 = (k0 + 32 < K) ? (k0 + 32) : 0;   // scalar select, branch-free wrap
        v16h ca0 = a0, ca1 = a1;
        // Prefetch A fragments for the next K-step (redundant wrap on last iter)
        a0 = frag_load_A(A, K, m0,      k1, lane);
        a1 = frag_load_A(A, K, m0 + 16, k1, lane);
#pragma unroll
        for (int j = 0; j < 4; ++j) {
            v16h cb = b;
            // Prefetch next B fragment: next N-tile, or first tile of next K-step
            int nn = (j < 3) ? (n0 + ((j + 1) << 4)) : n0;
            int nk = (j < 3) ? k0 : k1;
            b = frag_load_B(Bw, K, nn, nk, lane);
            acc[0][j] = __builtin_amdgcn_wmma_f32_16x16x32_f16(
                false, ca0, false, cb, (short)0, acc[0][j], false, false);
            acc[1][j] = __builtin_amdgcn_wmma_f32_16x16x32_f16(
                false, ca1, false, cb, (short)0, acc[1][j], false, false);
        }
    }
#pragma unroll
    for (int i = 0; i < 2; ++i)
#pragma unroll
        for (int j = 0; j < 4; ++j)
            frag_store_D(C, N, m0 + (i << 4), n0 + (j << 4), lane, acc[i][j]);
}

// ---------------------------------------------------------------------------
// f32 -> f16 conversion (weights, every call: deterministic, graph-safe)
// ---------------------------------------------------------------------------
__global__ void f32_to_f16(const float* __restrict__ s, _Float16* __restrict__ d, int n) {
    int i = blockIdx.x * blockDim.x + threadIdx.x;
    if (i < n) d[i] = (_Float16)s[i];
}

// x-proj weight (LAYERS x 48 x 512) -> f16 padded to (LAYERS x 64 x 512), rows 48..63 = 0
__global__ void xproj_to_f16_pad(const float* __restrict__ s, _Float16* __restrict__ d) {
    int idx = blockIdx.x * blockDim.x + threadIdx.x;
    if (idx >= LAYERS * NXP * DI) return;
    int k = idx & (DI - 1);
    int n = (idx >> 9) & (NXP - 1);
    int l = idx >> 15;                                  // / (64*512)
    d[idx] = (n < 48) ? (_Float16)s[((size_t)l * 48 + n) * DI + k] : (_Float16)0.f;
}

// ---------------------------------------------------------------------------
// Input projection: h[row, c] = sum_i x[row, i] * in_W[c, i]   (K = 15)
// ---------------------------------------------------------------------------
__global__ void input_proj(const float* __restrict__ x, const float* __restrict__ W,
                           float* __restrict__ h) {
    int idx = blockIdx.x * blockDim.x + threadIdx.x;
    if (idx >= BS * DM) return;
    int row = idx >> 8, c = idx & 255;
    const float* xr = x + (size_t)row * 15;
    const float* wr = W + (size_t)c * 15;
    float acc = 0.f;
#pragma unroll
    for (int i = 0; i < 15; ++i) acc += xr[i] * wr[i];
    h[idx] = acc;
}

// ---------------------------------------------------------------------------
// Residual accumulate + LayerNorm -> f16 row (one 256-thread block per row)
// ---------------------------------------------------------------------------
__global__ void __launch_bounds__(256)
ln_residual(const float* __restrict__ h, float* __restrict__ res,
            const float* __restrict__ nw, const float* __restrict__ nb,
            _Float16* __restrict__ hn, int first) {
    __shared__ float s1[256], s2[256];
    int row = blockIdx.x, c = threadIdx.x;
    size_t i = (size_t)row * DM + c;
    float r = first ? h[i] : (res[i] + h[i]);
    res[i] = r;
    s1[c] = r; s2[c] = r * r;
    __syncthreads();
    for (int off = 128; off > 0; off >>= 1) {
        if (c < off) { s1[c] += s1[c + off]; s2[c] += s2[c + off]; }
        __syncthreads();
    }
    float m = s1[0] * (1.f / DM);
    float v = s2[0] * (1.f / DM) - m * m;
    float inv = rsqrtf(v + 1e-5f);
    hn[i] = (_Float16)((r - m) * inv * nw[c] + nb[c]);
}

// ---------------------------------------------------------------------------
// Depthwise causal conv (k=4) + bias + SiLU over xz[:, 0:512]
// ---------------------------------------------------------------------------
__global__ void conv_silu(const float* __restrict__ xz, const float* __restrict__ cw,
                          const float* __restrict__ cb, float* __restrict__ xh32,
                          _Float16* __restrict__ xh16) {
    int idx = blockIdx.x * blockDim.x + threadIdx.x;
    if (idx >= BS * DI) return;
    int r = idx >> 9, d = idx & 511;
    int s = r & (SEQLEN - 1);
    const float* w = cw + d * 4;
    float acc = cb[d];
#pragma unroll
    for (int j = 0; j < 4; ++j) {
        if (s + j >= 3)                         // t = s-3+j >= 0
            acc += w[j] * xz[(size_t)(r + j - 3) * (2 * DI) + d];
    }
    float y = acc / (1.f + __expf(-acc));       // SiLU
    xh32[idx] = y;
    xh16[idx] = (_Float16)y;
}

// ---------------------------------------------------------------------------
// dt = softplus(dbl[:, :16] @ dtW^T + dtb)   (K=16 -> VALU, not worth WMMA)
// dbl row stride is NXP (64).
// ---------------------------------------------------------------------------
__global__ void dtproj_softplus(const float* __restrict__ dbl, const float* __restrict__ dtW,
                                const float* __restrict__ dtb, float* __restrict__ dtf) {
    int idx = blockIdx.x * blockDim.x + threadIdx.x;
    if (idx >= BS * DI) return;
    int row = idx >> 9, d = idx & 511;
    const float* x = dbl + (size_t)row * NXP;
    const float* w = dtW + d * DTRANK;
    float acc = dtb[d];
#pragma unroll
    for (int i = 0; i < DTRANK; ++i) acc += x[i] * w[i];
    dtf[idx] = (acc > 20.f) ? acc : log1pf(__expf(acc));
}

// ---------------------------------------------------------------------------
// Selective scan.  Thread = one (b, d, n) recurrence (32768 total).
// n lives in lanes 0..15 of each half-wave -> Sum_n via 4x shfl_xor.
// B/C rows staged in LDS in 64-step chunks.
// ---------------------------------------------------------------------------
__global__ void __launch_bounds__(512)
selective_scan(const float* __restrict__ dbl, const float* __restrict__ dtf,
               const float* __restrict__ xh, const float* __restrict__ A_log,
               float* __restrict__ y) {
    __shared__ float sBC[64 * 32];
    int tid = threadIdx.x;
    int n  = tid & 15;
    int d  = blockIdx.x * 32 + (tid >> 4);
    int b  = blockIdx.y;
    float Ac  = -__expf(A_log[(size_t)d * DSTATE + n]);
    float hst = 0.f;
    for (int t0 = 0; t0 < SEQLEN; t0 += 64) {
        __syncthreads();
        for (int i = tid; i < 64 * 32; i += 512) {
            int tt = i >> 5, c = i & 31;
            sBC[i] = dbl[(size_t)(b * SEQLEN + t0 + tt) * NXP + DTRANK + c];
        }
        __syncthreads();
        for (int tt = 0; tt < 64; ++tt) {
            size_t base = ((size_t)(b * SEQLEN) + t0 + tt) * DI + d;
            float dt = dtf[base];
            float xv = xh[base];
            float Bn = sBC[tt * 32 + n];
            float Cn = sBC[tt * 32 + 16 + n];
            hst = __expf(dt * Ac) * hst + dt * Bn * xv;
            float part = hst * Cn;
            part += __shfl_xor(part, 1, 32);
            part += __shfl_xor(part, 2, 32);
            part += __shfl_xor(part, 4, 32);
            part += __shfl_xor(part, 8, 32);
            if (n == 0) y[base] = part;
        }
    }
}

// ---------------------------------------------------------------------------
// Gate: yy = (y + Dp * xh) * silu(z)  -> f16 for out-proj
// ---------------------------------------------------------------------------
__global__ void gate_silu(const float* __restrict__ y, const float* __restrict__ xh,
                          const float* __restrict__ xz, const float* __restrict__ Dp,
                          _Float16* __restrict__ yy) {
    int idx = blockIdx.x * blockDim.x + threadIdx.x;
    if (idx >= BS * DI) return;
    int row = idx >> 9, d = idx & 511;
    float z = xz[(size_t)row * (2 * DI) + DI + d];
    float g = z / (1.f + __expf(-z));
    yy[idx] = (_Float16)((y[idx] + Dp[d] * xh[idx]) * g);
}

// ---------------------------------------------------------------------------
// Final: out[row] = h[row, :] . out_W[0, :]
// ---------------------------------------------------------------------------
__global__ void __launch_bounds__(256)
final_proj(const float* __restrict__ h, const float* __restrict__ W,
           float* __restrict__ out) {
    __shared__ float s1[256];
    int row = blockIdx.x, c = threadIdx.x;
    s1[c] = h[(size_t)row * DM + c] * W[c];
    __syncthreads();
    for (int off = 128; off > 0; off >>= 1) {
        if (c < off) s1[c] += s1[c + off];
        __syncthreads();
    }
    if (c == 0) out[row] = s1[0];
}

// ---------------------------------------------------------------------------
// Host driver
// ---------------------------------------------------------------------------
extern "C" void kernel_launch(void* const* d_in, const int* in_sizes, int n_in,
                              void* d_out, int out_size, void* d_ws, size_t ws_size,
                              hipStream_t stream) {
    (void)in_sizes; (void)n_in; (void)out_size; (void)ws_size;
    const float* x_src  = (const float*)d_in[0];
    const float* in_W   = (const float*)d_in[2];
    const float* norm_w = (const float*)d_in[3];
    const float* norm_b = (const float*)d_in[4];
    const float* inprojW= (const float*)d_in[5];
    const float* conv_w = (const float*)d_in[6];
    const float* conv_b = (const float*)d_in[7];
    const float* xprojW = (const float*)d_in[8];
    const float* dtW    = (const float*)d_in[9];
    const float* dtb    = (const float*)d_in[10];
    const float* A_log  = (const float*)d_in[11];
    const float* Dp     = (const float*)d_in[12];
    const float* oprojW = (const float*)d_in[13];
    const float* outW   = (const float*)d_in[14];
    float* out = (float*)d_out;

    char* ws = (char*)d_ws;
    auto carve = [&](size_t bytes) -> void* {
        void* p = (void*)ws;
        ws += (bytes + 255) & ~(size_t)255;
        return p;
    };
    float*    hbuf  = (float*)   carve((size_t)BS * DM * 4);
    float*    res   = (float*)   carve((size_t)BS * DM * 4);
    _Float16* hn16  = (_Float16*)carve((size_t)BS * DM * 2);
    float*    xz    = (float*)   carve((size_t)BS * 2 * DI * 4);
    float*    xh32  = (float*)   carve((size_t)BS * DI * 4);
    _Float16* xh16  = (_Float16*)carve((size_t)BS * DI * 2);
    float*    dbl   = (float*)   carve((size_t)BS * NXP * 4);
    float*    dtf   = (float*)   carve((size_t)BS * DI * 4);
    float*    yscan = (float*)   carve((size_t)BS * DI * 4);
    _Float16* yy16  = (_Float16*)carve((size_t)BS * DI * 2);
    _Float16* inW16 = (_Float16*)carve((size_t)LAYERS * 2 * DI * DM * 2);
    _Float16* xpW16 = (_Float16*)carve((size_t)LAYERS * NXP * DI * 2);
    _Float16* opW16 = (_Float16*)carve((size_t)LAYERS * DM * DI * 2);

    const int T256 = 256;
    // Weight conversions
    {
        int n1 = LAYERS * 2 * DI * DM;
        f32_to_f16<<<(n1 + T256 - 1) / T256, T256, 0, stream>>>(inprojW, inW16, n1);
        int n2 = LAYERS * NXP * DI;
        xproj_to_f16_pad<<<(n2 + T256 - 1) / T256, T256, 0, stream>>>(xprojW, xpW16);
        int n3 = LAYERS * DM * DI;
        f32_to_f16<<<(n3 + T256 - 1) / T256, T256, 0, stream>>>(oprojW, opW16, n3);
    }

    // h = x_src @ in_W^T
    input_proj<<<(BS * DM + T256 - 1) / T256, T256, 0, stream>>>(x_src, in_W, hbuf);

    int nEl = BS * DI;
    int gEl = (nEl + T256 - 1) / T256;

    // GEMM launch helper: waves/block = min(4, N/64), branch-free exact tiling
    auto launch_gemm = [&](const _Float16* A, const _Float16* Bw, float* C, int N, int K) {
        int wpb = (N / 64 < 4) ? (N / 64) : 4;
        dim3 grid(N / (64 * wpb), BS / 32);
        wmma_gemm_nt<<<grid, 32 * wpb, 0, stream>>>(A, Bw, C, BS, N, K);
    };

    for (int l = 0; l < LAYERS; ++l) {
        // res (+)= h ; hn = LN(res) -> f16
        ln_residual<<<BS, 256, 0, stream>>>(hbuf, res, norm_w + l * DM, norm_b + l * DM,
                                            hn16, l == 0);
        // xz = hn @ inW^T   (M=8192, N=1024, K=256)
        launch_gemm(hn16, inW16 + (size_t)l * 2 * DI * DM, xz, 2 * DI, DM);
        // depthwise conv + SiLU
        conv_silu<<<gEl, T256, 0, stream>>>(xz, conv_w + l * DI * 4, conv_b + l * DI,
                                            xh32, xh16);
        // dbl = xh @ xpW_pad^T  (M=8192, N=64, K=512)
        launch_gemm(xh16, xpW16 + (size_t)l * NXP * DI, dbl, NXP, DI);
        // dt = softplus(dt_in @ dtW^T + dtb)
        dtproj_softplus<<<gEl, T256, 0, stream>>>(dbl, dtW + (size_t)l * DI * DTRANK,
                                                  dtb + l * DI, dtf);
        // selective scan
        selective_scan<<<dim3(DI / 32, BATCH), 512, 0, stream>>>(
            dbl, dtf, xh32, A_log + (size_t)l * DI * DSTATE, yscan);
        // gate
        gate_silu<<<gEl, T256, 0, stream>>>(yscan, xh32, xz, Dp + l * DI, yy16);
        // h = yy @ opW^T    (M=8192, N=256, K=512)
        launch_gemm(yy16, opW16 + (size_t)l * DM * DI, hbuf, DM, DI);
    }

    // out = h @ out_W^T
    final_proj<<<BS, 256, 0, stream>>>(hbuf, outW, out);
}
